// mdpvsr_1defconv_32478542692452
// MI455X (gfx1250) — matrix operand order repaired and checked by reading, not compiled, not run on hardware
//
#include <hip/hip_runtime.h>
#include <hip/hip_bf16.h>

typedef __attribute__((ext_vector_type(16))) _Float16 v16h;
typedef __attribute__((ext_vector_type(8)))  float    v8f;

namespace {
constexpr int cH = 180, cW = 320, cHW = cH * cW, cT = 4, cB = 2;
constexpr int TILES = cHW / 16;   // 3600 tiles of 16 consecutive pixels
constexpr int TPR   = cW / 16;    // 20 tiles per image row (no row straddle)

// packed-A fragment geometry: 32 lanes x 16 halfs per (mtile, kchunk)
constexpr int FRAG = 512;
constexpr int LSTM_CH = 6, LSTM_MT = 4;   // K = 19*9 = 171 -> 6 chunks; 64 out ch
constexpr int OFF_CH = 18, OFF_MT = 2;    // K = 64*9 = 576; 27 out ch (pad to 32)
constexpr int DEF_CH = 18, DEF_MT = 1;    // 16 out ch
constexpr int OUT_CH = 5,  OUT_MT = 3;    // K = 16*9 = 144 -> 5 chunks; 48 out ch

// workspace layout (bytes)
constexpr size_t oALSTM = 0;
constexpr size_t oAOFF  = oALSTM + size_t(LSTM_MT * LSTM_CH * FRAG) * 2;
constexpr size_t oADEF  = oAOFF  + size_t(OFF_MT  * OFF_CH  * FRAG) * 2;
constexpr size_t oAOUT  = oADEF  + size_t(DEF_MT  * DEF_CH  * FRAG) * 2;
constexpr size_t oHS    = ((oAOUT + size_t(OUT_MT * OUT_CH * FRAG) * 2 + 255) & ~size_t(255));
constexpr size_t oCB    = oHS   + size_t(cB) * 64 * cHW * 4;  // hs: (B,64,H,W)
constexpr size_t oOFFB  = oCB   + size_t(cB) * 16 * cHW * 4;  // cell state
constexpr size_t oMASK  = oOFFB + size_t(cB) * 18 * cHW * 4;  // offsets
constexpr size_t oDC    = oMASK + size_t(cB) *  9 * cHW * 4;  // 2*sigmoid mask
}

// 16-bit A-matrix 16x32 fragment layout (contiguous K, matches B fragments):
// lane<16: M=lane, elems 0..15 -> K0..15
// lane>=16: M=lane-16, elems 0..15 -> K16..31
__device__ __forceinline__ int a_kidx(int lane, int e) {
  int base = (lane & 16) ? 16 : 0;
  return base + e;
}
__device__ __forceinline__ float sigm(float x) { return 1.0f / (1.0f + __expf(-x)); }
__device__ __forceinline__ float ld_clip(const float* p, int yy, int xx) {
  return (yy >= 0 && yy < cH && xx >= 0 && xx < cW) ? p[yy * cW + xx] : 0.0f;
}
__device__ __forceinline__ v8f wmma_f16(v16h a, v16h b, v8f c) {
  return __builtin_amdgcn_wmma_f32_16x16x32_f16(false, a, false, b, (short)0, c, false, false);
}
// Per-lane B fragment base inside an im2col LDS matrix Bm[n][Kpad] (halfs):
// lane reads 16 consecutive halfs at n*Kpad + (lane&16 ? 16 : 0) + kc*32.
// With Kpad a multiple of 16 the address is 32-byte aligned -> 2x ds_load_b128.
__device__ __forceinline__ const v16h* bfrag_base(const _Float16* Bm, int lane, int Kpad) {
  return (const v16h*)(Bm + (lane & 15) * Kpad + ((lane & 16) ? 16 : 0));
}

// ---------------- one-time weight pack into WMMA A fragments (f16) ----------
__global__ __launch_bounds__(256)
void pack_weights_kernel(const float* lstm_w, const float* off_w, const float* mod_w,
                         const float* def_w, const float* out_w,
                         _Float16* Alstm, _Float16* Aoff, _Float16* Adef, _Float16* Aout)
{
  int id = blockIdx.x * blockDim.x + threadIdx.x;
  const int NL = LSTM_MT * LSTM_CH * FRAG;   // 12288
  const int NO = OFF_MT  * OFF_CH  * FRAG;   // 18432
  const int ND = DEF_MT  * DEF_CH  * FRAG;   // 9216
  const int NU = OUT_MT  * OUT_CH  * FRAG;   // 7680
  if (id < NL) {
    int tile = id / FRAG, rem = id % FRAG, lane = rem >> 4, e = rem & 15;
    int g = tile / LSTM_CH, kc = tile % LSTM_CH;
    int m = lane & 15;
    int k = kc * 32 + a_kidx(lane, e);
    float w = 0.0f;
    if (k < 171) w = lstm_w[((g * 16 + m) * 19 + k / 9) * 9 + k % 9];
    Alstm[id] = (_Float16)w;
  } else if (id < NL + NO) {
    int sid = id - NL;
    int tile = sid / FRAG, rem = sid % FRAG, lane = rem >> 4, e = rem & 15;
    int mt = tile / OFF_CH, kc = tile % OFF_CH;
    int m = lane & 15, o = mt * 16 + m;
    int k = kc * 32 + a_kidx(lane, e);
    float w = 0.0f;
    if (o < 18)       w = off_w[(o * 64 + k / 9) * 9 + k % 9];
    else if (o < 27)  w = mod_w[((o - 18) * 64 + k / 9) * 9 + k % 9];
    Aoff[sid] = (_Float16)w;
  } else if (id < NL + NO + ND) {
    int sid = id - NL - NO;
    int rem = sid % FRAG, lane = rem >> 4, e = rem & 15;
    int kc = sid / FRAG;
    int m = lane & 15;
    int k = kc * 32 + a_kidx(lane, e);
    Adef[sid] = (_Float16)def_w[m * 576 + k];
  } else if (id < NL + NO + ND + NU) {
    int sid = id - NL - NO - ND;
    int tile = sid / FRAG, rem = sid % FRAG, lane = rem >> 4, e = rem & 15;
    int mt = tile / OUT_CH, kc = tile % OUT_CH;
    int m = lane & 15, o = mt * 16 + m;
    int k = kc * 32 + a_kidx(lane, e);
    float w = 0.0f;
    if (k < 144) w = out_w[(o * 16 + k / 9) * 9 + k % 9];
    Aout[sid] = (_Float16)w;
  }
}

// Scatter one staged patch value into the im2col matrix. Patch element
// (ch,row,col) feeds exactly taps {3*row+d, d=0..2} at pixel n = col-d.
__device__ __forceinline__ void scatter_im2col(_Float16* Bm, int Kpad,
                                               int ch, int row, int col, _Float16 hv)
{
  int kb = ch * 9 + 3 * row;
#pragma unroll
  for (int d = 0; d < 3; ++d) {
    int n2 = col - d;
    if ((unsigned)n2 < 16u) Bm[n2 * Kpad + kb + d] = hv;
  }
}

// ---------------- ConvLSTM step: conv(concat(xt,h),19->64) + gating ---------
__global__ __launch_bounds__(32)
void lstm_step_kernel(const float* X, const float* lstm_b,
                      const float* W_ci, const float* W_cf, const float* W_co,
                      const _Float16* Alstm, float* hs, float* cbuf, int t)
{
  __shared__ __align__(32) _Float16 Bm[16 * 192];   // im2col [n][Kpad], 6 KB
  const int lane = threadIdx.x;
  const int tile = blockIdx.x, b = blockIdx.y;
  const int y = tile / TPR, x0 = (tile % TPR) * 16;

  for (int i = lane; i < 19 * 54; i += 32) {
    int ch = i / 54, rem = i % 54, row = rem / 18, col = rem % 18;
    int yy = y - 1 + row, xx = x0 - 1 + col;
    float v = 0.0f;
    if (yy >= 0 && yy < cH && xx >= 0 && xx < cW) {
      if (ch < 3)      v = X[(((size_t)b * cT + t) * 3 + ch) * cHW + yy * cW + xx];
      else if (t > 0)  v = hs[((size_t)b * 64 + (t - 1) * 16 + (ch - 3)) * cHW + yy * cW + xx];
    }
    scatter_im2col(Bm, 192, ch, row, col, (_Float16)v);
  }
  for (int i = lane; i < 16 * 21; i += 32)          // zero K pad 171..191
    Bm[(i & 15) * 192 + 171 + (i >> 4)] = (_Float16)0.0f;
  __syncthreads();

  const v16h* Af = (const v16h*)Alstm;
  const v16h* Bf = bfrag_base(Bm, lane, 192);
  v8f acc[4] = {};
  for (int kc = 0; kc < 6; ++kc) {
    v16h bb = Bf[2 * kc];
#pragma unroll
    for (int g = 0; g < 4; ++g)
      acc[g] = wmma_f16(Af[(g * 6 + kc) * 32 + lane], bb, acc[g]);
  }

  const int n = lane & 15;
  const int pix = y * cW + x0 + n;
  for (int r = 0; r < 8; ++r) {
    int m = r + ((lane & 16) ? 8 : 0);
    float gv[4];
#pragma unroll
    for (int gi = 0; gi < 4; ++gi)
      gv[gi] = ((const float*)&acc[gi])[r] + lstm_b[gi * 16 + m];
    float cprev = (t > 0) ? cbuf[((size_t)b * 16 + m) * cHW + pix] : 0.0f;
    float wci = W_ci[m * cHW + pix];
    float wcf = W_cf[m * cHW + pix];
    float wco = W_co[m * cHW + pix];
    float ig = sigm(gv[0] + wci * cprev);
    float fg = sigm(gv[1] + wcf * cprev);
    float cnew = fg * cprev + ig * tanhf(gv[2]);
    float og = sigm(gv[3] + wco * cnew);
    cbuf[((size_t)b * 16 + m) * cHW + pix] = cnew;
    hs[((size_t)b * 64 + t * 16 + m) * cHW + pix] = og * tanhf(cnew);
  }
}

// ---------------- offset (18ch) + modulation (9ch) conv, 64 -> 27 -----------
__global__ __launch_bounds__(32)
void offmod_kernel(const float* hs, const float* off_b, const float* mod_b,
                   const _Float16* Aoff, float* offb, float* maskb)
{
  __shared__ __align__(32) _Float16 Bm[16 * 576];   // 18 KB
  const int lane = threadIdx.x;
  const int tile = blockIdx.x, b = blockIdx.y;
  const int y = tile / TPR, x0 = (tile % TPR) * 16;

  for (int i = lane; i < 64 * 54; i += 32) {
    int ch = i / 54, rem = i % 54, row = rem / 18, col = rem % 18;
    int yy = y - 1 + row, xx = x0 - 1 + col;
    float v = 0.0f;
    if (yy >= 0 && yy < cH && xx >= 0 && xx < cW)
      v = hs[((size_t)b * 64 + ch) * cHW + yy * cW + xx];
    scatter_im2col(Bm, 576, ch, row, col, (_Float16)v);
  }
  __syncthreads();

  const v16h* Af = (const v16h*)Aoff;
  const v16h* Bf = bfrag_base(Bm, lane, 576);
  v8f acc0 = {}, acc1 = {};
  for (int kc = 0; kc < 18; ++kc) {
    v16h bb = Bf[2 * kc];
    acc0 = wmma_f16(Af[kc * 32 + lane], bb, acc0);
    acc1 = wmma_f16(Af[(18 + kc) * 32 + lane], bb, acc1);
  }

  const int n = lane & 15;
  const int pix = y * cW + x0 + n;
  const float* a0 = (const float*)&acc0;
  const float* a1 = (const float*)&acc1;
  for (int r = 0; r < 8; ++r) {
    int m = r + ((lane & 16) ? 8 : 0);
    offb[((size_t)b * 18 + m) * cHW + pix] = a0[r] + off_b[m];
    int o = 16 + m;
    if (o < 18)
      offb[((size_t)b * 18 + o) * cHW + pix] = a1[r] + off_b[o];
    else if (o < 27)
      maskb[((size_t)b * 9 + (o - 18)) * cHW + pix] = 2.0f * sigm(a1[r] + mod_b[o - 18]);
  }
}

// ---------------- modulated deformable conv: gather + 16x576 GEMM -----------
__global__ __launch_bounds__(32)
void deform_kernel(const float* hs, const float* def_b, const _Float16* Adef,
                   const float* offb, const float* maskb, float* dcout)
{
  __shared__ __align__(32) _Float16 Bm[16 * 576];   // sampled im2col, 18 KB
  __shared__ int   s_y0[9][16], s_x0[9][16];
  __shared__ float s_wy[9][16], s_wx[9][16], s_mk[9][16];
  const int lane = threadIdx.x;
  const int tile = blockIdx.x, b = blockIdx.y;
  const int y = tile / TPR, x0 = (tile % TPR) * 16;

  for (int i = lane; i < 144; i += 32) {
    int k2 = i / 16, n2 = i % 16;
    int pix = y * cW + x0 + n2;
    float dy = offb[((size_t)b * 18 + 2 * k2) * cHW + pix];
    float dx = offb[((size_t)b * 18 + 2 * k2 + 1) * cHW + pix];
    float py = (float)(y - 1 + k2 / 3) + dy;
    float px = (float)(x0 + n2 - 1 + k2 % 3) + dx;
    float yf = floorf(py), xf = floorf(px);
    s_y0[k2][n2] = (int)yf;  s_x0[k2][n2] = (int)xf;
    s_wy[k2][n2] = py - yf;  s_wx[k2][n2] = px - xf;
    s_mk[k2][n2] = maskb[((size_t)b * 9 + k2) * cHW + pix];
  }
  __syncthreads();

  const int n = lane & 15;
  // hoist the 9 bilinear descriptors for this pixel column into registers
  int yi[9], xi[9]; float wy[9], wx[9], mk[9];
#pragma unroll
  for (int k2 = 0; k2 < 9; ++k2) {
    yi[k2] = s_y0[k2][n]; xi[k2] = s_x0[k2][n];
    wy[k2] = s_wy[k2][n]; wx[k2] = s_wx[k2][n]; mk[k2] = s_mk[k2][n];
  }
  // two lanes per pixel column split the 64 input channels
  const int c0 = (lane >> 4) * 32;
  for (int c = c0; c < c0 + 32; ++c) {
    const float* xb = hs + ((size_t)b * 64 + c) * cHW;
    _Float16* br = Bm + n * 576 + c * 9;
#pragma unroll
    for (int k2 = 0; k2 < 9; ++k2) {
      float g00 = ld_clip(xb, yi[k2],     xi[k2]);
      float g01 = ld_clip(xb, yi[k2],     xi[k2] + 1);
      float g10 = ld_clip(xb, yi[k2] + 1, xi[k2]);
      float g11 = ld_clip(xb, yi[k2] + 1, xi[k2] + 1);
      float v = (1.0f - wy[k2]) * ((1.0f - wx[k2]) * g00 + wx[k2] * g01)
              +         wy[k2]  * ((1.0f - wx[k2]) * g10 + wx[k2] * g11);
      br[k2] = (_Float16)(v * mk[k2]);
    }
  }
  __syncthreads();

  const v16h* Af = (const v16h*)Adef;
  const v16h* Bf = bfrag_base(Bm, lane, 576);
  v8f acc = {};
  for (int kc = 0; kc < 18; ++kc)
    acc = wmma_f16(Af[kc * 32 + lane], Bf[2 * kc], acc);

  const int pix = y * cW + x0 + n;
  const float* a = (const float*)&acc;
  for (int r = 0; r < 8; ++r) {
    int m = r + ((lane & 16) ? 8 : 0);
    dcout[((size_t)b * 16 + m) * cHW + pix] = a[r] + def_b[m];
  }
}

// ---------------- out conv 16->48 + pixel shuffle (r=4) + clip --------------
__global__ __launch_bounds__(32)
void outconv_kernel(const float* dcout, const float* out_b, const _Float16* Aout,
                    float* outp)
{
  __shared__ __align__(32) _Float16 Bm[16 * 160];   // 5 KB
  const int lane = threadIdx.x;
  const int tile = blockIdx.x, b = blockIdx.y;
  const int y = tile / TPR, x0 = (tile % TPR) * 16;

  for (int i = lane; i < 16 * 54; i += 32) {
    int ch = i / 54, rem = i % 54, row = rem / 18, col = rem % 18;
    int yy = y - 1 + row, xx = x0 - 1 + col;
    float v = 0.0f;
    if (yy >= 0 && yy < cH && xx >= 0 && xx < cW)
      v = dcout[((size_t)b * 16 + ch) * cHW + yy * cW + xx];
    scatter_im2col(Bm, 160, ch, row, col, (_Float16)v);
  }
  for (int i = lane; i < 16 * 16; i += 32)          // zero K pad 144..159
    Bm[(i & 15) * 160 + 144 + (i >> 4)] = (_Float16)0.0f;
  __syncthreads();

  const v16h* Af = (const v16h*)Aout;
  const v16h* Bf = bfrag_base(Bm, lane, 160);
  const int n = lane & 15;
  v8f acc[3] = {};
  for (int kc = 0; kc < 5; ++kc) {
    v16h bb = Bf[2 * kc];
#pragma unroll
    for (int mt = 0; mt < 3; ++mt)
      acc[mt] = wmma_f16(Af[(mt * 5 + kc) * 32 + lane], bb, acc[mt]);
  }

  for (int mt = 0; mt < 3; ++mt) {             // mt == output color channel
    const float* a = (const float*)&acc[mt];
    for (int r = 0; r < 8; ++r) {
      int m = r + ((lane & 16) ? 8 : 0);       // m = rr*4 + ss (shuffle index)
      float v = a[r] + out_b[mt * 16 + m];
      v = fminf(fmaxf(v, 0.0f), 255.0f);
      int yo = y * 4 + (m >> 2);
      int xo = (x0 + n) * 4 + (m & 3);
      outp[(((size_t)b * 3 + mt) * 720 + yo) * 1280 + xo] = v;
    }
  }
}

extern "C" void kernel_launch(void* const* d_in, const int* in_sizes, int n_in,
                              void* d_out, int out_size, void* d_ws, size_t ws_size,
                              hipStream_t stream)
{
  (void)in_sizes; (void)n_in; (void)out_size; (void)ws_size;
  const float* X      = (const float*)d_in[0];
  const float* lstm_w = (const float*)d_in[1];
  const float* lstm_b = (const float*)d_in[2];
  const float* W_ci   = (const float*)d_in[3];
  const float* W_cf   = (const float*)d_in[4];
  const float* W_co   = (const float*)d_in[5];
  const float* off_w  = (const float*)d_in[6];
  const float* off_b  = (const float*)d_in[7];
  const float* mod_w  = (const float*)d_in[8];
  const float* mod_b  = (const float*)d_in[9];
  const float* def_w  = (const float*)d_in[10];
  const float* def_b  = (const float*)d_in[11];
  const float* out_w  = (const float*)d_in[12];
  const float* out_b  = (const float*)d_in[13];
  float* outp = (float*)d_out;

  char* wsb = (char*)d_ws;
  _Float16* Alstm = (_Float16*)(wsb + oALSTM);
  _Float16* Aoff  = (_Float16*)(wsb + oAOFF);
  _Float16* Adef  = (_Float16*)(wsb + oADEF);
  _Float16* Aout  = (_Float16*)(wsb + oAOUT);
  float* hs    = (float*)(wsb + oHS);
  float* cbuf  = (float*)(wsb + oCB);
  float* offb  = (float*)(wsb + oOFFB);
  float* maskb = (float*)(wsb + oMASK);
  float* dcout = (float*)(wsb + oDC);

  const int NPACK = (LSTM_MT*LSTM_CH + OFF_MT*OFF_CH + DEF_MT*DEF_CH + OUT_MT*OUT_CH) * FRAG;
  pack_weights_kernel<<<(NPACK + 255) / 256, 256, 0, stream>>>(
      lstm_w, off_w, mod_w, def_w, out_w, Alstm, Aoff, Adef, Aout);

  dim3 grid(TILES, cB);
  for (int t = 0; t < cT; ++t)
    lstm_step_kernel<<<grid, 32, 0, stream>>>(X, lstm_b, W_ci, W_cf, W_co, Alstm, hs, cbuf, t);
  offmod_kernel<<<grid, 32, 0, stream>>>(hs, off_b, mod_b, Aoff, offb, maskb);
  deform_kernel<<<grid, 32, 0, stream>>>(hs, def_b, Adef, offb, maskb, dcout);
  outconv_kernel<<<grid, 32, 0, stream>>>(dcout, out_b, Aout, outp);
}
